// GRU_35622458753791
// MI455X (gfx1250) — compile-verified, matrix operand order checked by reading, not run on hardware
//
#include <hip/hip_runtime.h>
#include <cmath>

#define NHEADS   8
#define HDIM     64
#define STATE    512          // NHEADS * HDIM
#define BATCH    4
#define SEQ      4096
#define DMODEL   1024
#define PROJW    (4 * STATE)  // 2048

typedef __attribute__((ext_vector_type(16))) __bf16 v16bf;
typedef __attribute__((ext_vector_type(8)))  __bf16 v8bf;
typedef __attribute__((ext_vector_type(8)))  float  v8f;

// -------------------------------------------------------------------------
// GEMM: C[M,N] = A[M,K] * B[N,K]^T   (A,B fp32 in memory; bf16 WMMA compute,
// fp32 accumulate).  BM=128, BN=64, BK=32, 256 threads = 8 waves (4x2),
// each wave owns a 32x32 output tile = 2x2 v_wmma_f32_16x16x32_bf16 tiles.
// -------------------------------------------------------------------------
#define BM 128
#define BN 64
#define BK 32

__global__ __launch_bounds__(256)
void gemm_bf16_wmma(const float* __restrict__ A, const float* __restrict__ Bm,
                    float* __restrict__ C, int M, int N, int K) {
    __shared__ __align__(16) __bf16 ldsA[BM * BK];
    __shared__ __align__(16) __bf16 ldsB[BN * BK];

    const int tid  = threadIdx.x;
    const int lane = tid & 31;
    const int wid  = tid >> 5;
    const int lr   = lane & 15;   // row/col within 16x16 tile
    const int hi   = lane >> 4;   // K-half selector (ISA A/B frag layout)

    const int waveN = wid & 1;    // 2 wave columns
    const int waveM = wid >> 1;   // 4 wave rows
    const int blockN0 = blockIdx.x * BN;
    const int blockM0 = blockIdx.y * BM;

    // global->LDS staging map
    const int arow = tid >> 1;            // 0..127
    const int acol = (tid & 1) * 16;      // 0 or 16
    const int brow = tid >> 2;            // 0..63
    const int bcol = (tid & 3) * 8;       // 0,8,16,24

    v8f acc[2][2];
    #pragma unroll
    for (int i = 0; i < 2; ++i)
        #pragma unroll
        for (int j = 0; j < 2; ++j)
            #pragma unroll
            for (int e = 0; e < 8; ++e) acc[i][j][e] = 0.0f;

    union Frag { v16bf v; v8bf h[2]; };

    for (int k0 = 0; k0 < K; k0 += BK) {
        __syncthreads();   // protect LDS from previous iteration's readers

        // stage A tile: 128x32 fp32 -> bf16
        {
            const float4* ga = (const float4*)(A + (size_t)(blockM0 + arow) * K + k0 + acol);
            float4 f0 = ga[0], f1 = ga[1], f2 = ga[2], f3 = ga[3];
            v8bf p0, p1;
            p0[0]=(__bf16)f0.x; p0[1]=(__bf16)f0.y; p0[2]=(__bf16)f0.z; p0[3]=(__bf16)f0.w;
            p0[4]=(__bf16)f1.x; p0[5]=(__bf16)f1.y; p0[6]=(__bf16)f1.z; p0[7]=(__bf16)f1.w;
            p1[0]=(__bf16)f2.x; p1[1]=(__bf16)f2.y; p1[2]=(__bf16)f2.z; p1[3]=(__bf16)f2.w;
            p1[4]=(__bf16)f3.x; p1[5]=(__bf16)f3.y; p1[6]=(__bf16)f3.z; p1[7]=(__bf16)f3.w;
            *(v8bf*)&ldsA[arow * BK + acol]     = p0;
            *(v8bf*)&ldsA[arow * BK + acol + 8] = p1;
        }
        // stage B tile: 64x32 fp32 -> bf16  (rows of B = rows of W = columns of B^T)
        {
            const float4* gb = (const float4*)(Bm + (size_t)(blockN0 + brow) * K + k0 + bcol);
            float4 f0 = gb[0], f1 = gb[1];
            v8bf p;
            p[0]=(__bf16)f0.x; p[1]=(__bf16)f0.y; p[2]=(__bf16)f0.z; p[3]=(__bf16)f0.w;
            p[4]=(__bf16)f1.x; p[5]=(__bf16)f1.y; p[6]=(__bf16)f1.z; p[7]=(__bf16)f1.w;
            *(v8bf*)&ldsB[brow * BK + bcol] = p;
        }
        __syncthreads();

        // fragments: lane<16 -> K {0..7,16..23}; lane>=16 -> K {8..15,24..31}
        Frag af[2], bf2[2];
        #pragma unroll
        for (int i = 0; i < 2; ++i) {
            const __bf16* base = &ldsA[(waveM * 32 + i * 16 + lr) * BK + hi * 8];
            af[i].h[0] = *(const v8bf*)(base);
            af[i].h[1] = *(const v8bf*)(base + 16);
        }
        #pragma unroll
        for (int j = 0; j < 2; ++j) {
            const __bf16* base = &ldsB[(waveN * 32 + j * 16 + lr) * BK + hi * 8];
            bf2[j].h[0] = *(const v8bf*)(base);
            bf2[j].h[1] = *(const v8bf*)(base + 16);
        }
        #pragma unroll
        for (int i = 0; i < 2; ++i)
            #pragma unroll
            for (int j = 0; j < 2; ++j)
                acc[i][j] = __builtin_amdgcn_wmma_f32_16x16x32_bf16(
                    false, af[i].v, false, bf2[j].v, (short)0, acc[i][j], false, false);
    }

    // epilogue: v8f element e -> row = hi*8 + e, col = lr (within 16x16 tile)
    #pragma unroll
    for (int i = 0; i < 2; ++i)
        #pragma unroll
        for (int j = 0; j < 2; ++j) {
            int r0 = blockM0 + waveM * 32 + i * 16 + hi * 8;
            int c0 = blockN0 + waveN * 32 + j * 16 + lr;
            #pragma unroll
            for (int e = 0; e < 8; ++e)
                C[(size_t)(r0 + e) * N + c0] = acc[i][j][e];
        }
}

// -------------------------------------------------------------------------
// Depthwise causal conv (K=4) + SiLU on the xi slice of proj -> ybuf
// -------------------------------------------------------------------------
__global__ __launch_bounds__(256)
void conv_silu_kernel(const float* __restrict__ proj, const float* __restrict__ cw,
                      float* __restrict__ out) {
    int idx = blockIdx.x * 256 + threadIdx.x;          // over B*S*STATE
    if (idx >= BATCH * SEQ * STATE) return;
    int c  = idx & (STATE - 1);
    int s  = (idx >> 9) & (SEQ - 1);
    int bs = idx >> 9;                                  // b*SEQ + s
    float w0 = cw[c * 4 + 0], w1 = cw[c * 4 + 1], w2 = cw[c * 4 + 2], w3 = cw[c * 4 + 3];
    const float* p = proj + (size_t)bs * PROJW + c;
    float acc = w3 * p[0];
    if (s >= 1) acc += w2 * p[-(ptrdiff_t)PROJW];
    if (s >= 2) acc += w1 * p[-(ptrdiff_t)(2 * PROJW)];
    if (s >= 3) acc += w0 * p[-(ptrdiff_t)(3 * PROJW)];
    out[idx] = acc / (1.0f + __expf(-acc));             // silu
}

// -------------------------------------------------------------------------
// Sequential gated scan. One block per (batch, head): 32 blocks, 128 threads.
// Weights in LDS (48 KB); column access W[j*64+t] -> lane t hits bank t
// (conflict-free); h[j] is an LDS broadcast. ybuf holds xi in, h out (same
// slot, consumed then overwritten within a step).
// -------------------------------------------------------------------------
__global__ __launch_bounds__(128)
void scan_kernel(const float* __restrict__ proj, const float* __restrict__ sw,
                 float* __restrict__ ybuf) {
    __shared__ float Wz[HDIM * HDIM];
    __shared__ float Wf[HDIM * HDIM];
    __shared__ float Wr[HDIM * HDIM];
    __shared__ float h[HDIM];
    __shared__ float hr[HDIM];

    const int t = threadIdx.x;
    const int b = blockIdx.x >> 3;
    const int n = blockIdx.x & 7;

    const float* gW  = sw + (size_t)n * (HDIM * HDIM);
    const float* gWf = sw + (size_t)(NHEADS + n) * (HDIM * HDIM);
    const float* gWr = sw + (size_t)(2 * NHEADS + n) * (HDIM * HDIM);
    for (int i = t; i < HDIM * HDIM; i += 128) { Wz[i] = gW[i]; Wf[i] = gWf[i]; Wr[i] = gWr[i]; }
    if (t < HDIM) { h[t] = 0.0f; hr[t] = 0.0f; }
    __syncthreads();

    const int  tl  = t & 63;
    const bool isF = t < 64;
    const float* Wg = isF ? Wf : Wr;
    const size_t gateOff = (size_t)(isF ? STATE : 2 * STATE) + (size_t)n * HDIM + tl;

    for (int s = 0; s < SEQ; ++s) {
        const size_t prow = ((size_t)b * SEQ + s) * PROJW;
        const size_t yrow = ((size_t)b * SEQ + s) * STATE + (size_t)n * HDIM;

        // phase A: f (threads 0-63) and r (threads 64-127)
        float acc = 0.0f;
        #pragma unroll 8
        for (int j = 0; j < HDIM; ++j) acc += h[j] * Wg[j * HDIM + tl];
        float xg   = proj[prow + gateOff];
        float gate = 1.0f / (1.0f + __expf(-(acc + xg)));
        if (!isF) hr[tl] = h[tl] * gate;
        __syncthreads();

        // phase B: z and h update (threads 0-63)
        float hn = 0.0f;
        if (isF) {
            float accz = 0.0f;
            #pragma unroll 8
            for (int j = 0; j < HDIM; ++j) accz += hr[j] * Wz[j * HDIM + tl];
            float xi = ybuf[yrow + tl];
            float z  = tanhf(accz + xi);
            hn = gate * h[tl] + (1.0f - gate) * z;
        }
        __syncthreads();
        if (isF) { h[tl] = hn; ybuf[yrow + tl] = hn; }
        __syncthreads();
    }
}

// -------------------------------------------------------------------------
// y = rmsnorm(y * silu(g)) * norm_w   (one block per (b,s) row)
// -------------------------------------------------------------------------
__global__ __launch_bounds__(128)
void gate_norm_kernel(const float* __restrict__ proj, const float* __restrict__ norm_w,
                      float* __restrict__ ybuf) {
    __shared__ float red[128];
    const int row = blockIdx.x;              // b*SEQ + s
    const int t   = threadIdx.x;
    const float* g = proj + (size_t)row * PROJW + 3 * STATE;
    float* y = ybuf + (size_t)row * STATE;

    float v[4];
    float ss = 0.0f;
    #pragma unroll
    for (int u = 0; u < 4; ++u) {
        int c = u * 128 + t;
        float gv = g[c];
        float yy = y[c] * (gv / (1.0f + __expf(-gv)));
        v[u] = yy;
        ss += yy * yy;
    }
    red[t] = ss;
    __syncthreads();
    for (int off = 64; off > 0; off >>= 1) {
        if (t < off) red[t] += red[t + off];
        __syncthreads();
    }
    float rs = rsqrtf(red[0] * (1.0f / STATE) + 1e-6f);
    #pragma unroll
    for (int u = 0; u < 4; ++u) {
        int c = u * 128 + t;
        y[c] = v[u] * rs * norm_w[c];
    }
}

// -------------------------------------------------------------------------
extern "C" void kernel_launch(void* const* d_in, const int* in_sizes, int n_in,
                              void* d_out, int out_size, void* d_ws, size_t ws_size,
                              hipStream_t stream) {
    const float* x     = (const float*)d_in[0];
    const float* w_in  = (const float*)d_in[1];
    const float* cw    = (const float*)d_in[2];
    const float* sw    = (const float*)d_in[3];
    const float* nw    = (const float*)d_in[4];
    const float* w_out = (const float*)d_in[5];
    float* out = (float*)d_out;

    float* proj = (float*)d_ws;                                   // 16384 x 2048 fp32
    float* ybuf = proj + (size_t)BATCH * SEQ * PROJW;             // 16384 x 512  fp32

    const int Mrows = BATCH * SEQ;  // 16384

    // 1) proj = x @ w_in^T   (68.7 GFLOP, bf16 WMMA)
    dim3 g1(PROJW / BN, Mrows / BM);
    gemm_bf16_wmma<<<g1, 256, 0, stream>>>(x, w_in, proj, Mrows, PROJW, DMODEL);

    // 2) depthwise conv + silu -> ybuf
    int total = BATCH * SEQ * STATE;
    conv_silu_kernel<<<(total + 255) / 256, 256, 0, stream>>>(proj, cw, ybuf);

    // 3) sequential scan (in-place on ybuf)
    scan_kernel<<<BATCH * NHEADS, 128, 0, stream>>>(proj, sw, ybuf);

    // 4) output gate + rmsnorm (in-place on ybuf)
    gate_norm_kernel<<<Mrows, 128, 0, stream>>>(proj, nw, ybuf);

    // 5) out = ybuf @ w_out^T  (17.2 GFLOP, bf16 WMMA)
    dim3 g5(DMODEL / BN, Mrows / BM);
    gemm_bf16_wmma<<<g5, 256, 0, stream>>>(ybuf, w_out, out, Mrows, DMODEL, STATE);
}
